// LawModel_23270132810413
// MI455X (gfx1250) — matrix operand order repaired and checked by reading, not compiled, not run on hardware
//
#include <hip/hip_runtime.h>
#include <hip/hip_bf16.h>

// ---------------- problem constants ----------------
#define Vn   50000
#define En   256
#define Hn   256
#define H3n  768
#define MLPn 512
#define Cn   183
#define Bn   64
#define Sn   512
#define H2n  512

// ---------------- WMMA vector types ----------------
typedef __bf16 bf16x16 __attribute__((ext_vector_type(16)));
typedef __bf16 bf16x8  __attribute__((ext_vector_type(8)));
typedef float  f32x8   __attribute__((ext_vector_type(8)));

__device__ __forceinline__ unsigned short f2bf_u16(float x) {
    __bf16 b = (__bf16)x;
    union { __bf16 b; unsigned short u; } cvt; cvt.b = b; return cvt.u;
}
__device__ __forceinline__ float bfu2f(unsigned short u) {
    return __uint_as_float(((unsigned)u) << 16);
}

// A-matrix (16x32 bf16) per-lane load, per ISA layout:
//   half=0: K = kb+[0..7] and kb+16+[0..7]; half=1: K = kb+8+[0..7] and kb+24+[0..7]
__device__ __forceinline__ bf16x16 load_a_bf16(const __bf16* rowptr, int kb, int half) {
    bf16x16 r;
    const bf16x8* p0 = (const bf16x8*)(rowptr + kb + half * 8);
    const bf16x8* p1 = (const bf16x8*)(rowptr + kb + 16 + half * 8);
    bf16x8 lo = *p0, hi = *p1;
#pragma unroll
    for (int i = 0; i < 8; ++i) { r[i] = lo[i]; r[8 + i] = hi[i]; }
    return r;
}

// Same A layout sourced from f32 rows (embedding), converted in-register.
__device__ __forceinline__ bf16x16 load_a_f32(const float* rowptr, int kb, int half) {
    bf16x16 r;
#pragma unroll
    for (int c = 0; c < 2; ++c) {
        const float4* p = (const float4*)(rowptr + kb + c * 16 + half * 8);
        float4 f0 = p[0], f1 = p[1];
        r[c * 8 + 0] = (__bf16)f0.x; r[c * 8 + 1] = (__bf16)f0.y;
        r[c * 8 + 2] = (__bf16)f0.z; r[c * 8 + 3] = (__bf16)f0.w;
        r[c * 8 + 4] = (__bf16)f1.x; r[c * 8 + 5] = (__bf16)f1.y;
        r[c * 8 + 6] = (__bf16)f1.z; r[c * 8 + 7] = (__bf16)f1.w;
    }
    return r;
}

// B-matrix (32x16 bf16): lane holds column N=lane&15; half=0: K = kb+[0..7], kb+16+[0..7]; half=1: K = kb+8+[0..7], kb+24+[0..7].
__device__ __forceinline__ bf16x16 load_b_bf16(const __bf16* rowptr, int kb, int half) {
    bf16x16 r;
    const bf16x8* p0 = (const bf16x8*)(rowptr + kb + half * 8);
    const bf16x8* p1 = (const bf16x8*)(rowptr + kb + 16 + half * 8);
    bf16x8 lo = *p0, hi = *p1;
#pragma unroll
    for (int i = 0; i < 8; ++i) { r[i] = lo[i]; r[8 + i] = hi[i]; }
    return r;
}

__device__ __forceinline__ f32x8 wmma_bf16(bf16x16 a, bf16x16 b, f32x8 c) {
    return __builtin_amdgcn_wmma_f32_16x16x32_bf16(false, a, false, b, (short)0, c, false, false);
}

__device__ __forceinline__ float dev_sigmoid(float x) { return 1.f / (1.f + __expf(-x)); }

// ---------------- kernel 0: f32 -> bf16 weight conversion ----------------
__global__ void cvt_bf16_kernel(const float* wihf, const float* wihb,
                                const float* whhf, const float* whhb,
                                unsigned short* wih, unsigned short* whh) {
    const int NM = H3n * Hn; // 196608 per matrix
    int id = blockIdx.x * blockDim.x + threadIdx.x;
    if (id >= 4 * NM) return;
    int mat = id / NM, r = id % NM;
    if      (mat == 0) wih[r]      = f2bf_u16(wihf[r]);
    else if (mat == 1) wih[NM + r] = f2bf_u16(wihb[r]);
    else if (mat == 2) whh[r]      = f2bf_u16(whhf[r]);
    else               whh[NM + r] = f2bf_u16(whhb[r]);
}

// ---------------- kernel 1: gx = gather(emb) @ W_ih^T + b_ih, stored bf16 ----------------
// gx layout: [dir][S*B][3H] bf16, row r = s*B + b.
__global__ void gx_wmma_kernel(const int* __restrict__ facts, const float* __restrict__ emb,
                               const unsigned short* __restrict__ wih_bf,
                               const float* __restrict__ bih_f, const float* __restrict__ bih_b,
                               unsigned short* __restrict__ gx) {
    const int MT = (Sn * Bn) / 16;   // 2048
    const int NT = H3n / 16;         // 48
    const int TILES = MT * NT;       // 98304 per dir
    int wave = threadIdx.x >> 5;
    int lane = threadIdx.x & 31;
    int idx = lane & 15, half = lane >> 4;

    long tileId = (long)blockIdx.x * 8 + wave;
    int dir = tileId >= TILES;
    int t = (int)(tileId - (long)dir * TILES);
    int mt = t / NT, nt = t % NT;

    int r = mt * 16 + idx;          // token row for this lane
    int b = r & (Bn - 1);
    int s = r >> 6;
    int tok = facts[b * Sn + s];
    const float* arow = emb + (size_t)tok * En;

    const __bf16* wbase = (const __bf16*)(wih_bf) + (size_t)dir * H3n * Hn;
    const __bf16* brow = wbase + (size_t)(nt * 16 + idx) * Hn;

    f32x8 acc = {0.f, 0.f, 0.f, 0.f, 0.f, 0.f, 0.f, 0.f};
#pragma unroll
    for (int kk = 0; kk < 8; ++kk) {
        int kb = kk * 32;
        bf16x16 a = load_a_f32(arow, kb, half);
        bf16x16 bb = load_b_bf16(brow, kb, half);
        acc = wmma_bf16(a, bb, acc);
    }

    const float* bih = dir ? bih_b : bih_f;
    float bias = bih[nt * 16 + idx];
    unsigned short* gxd = gx + (size_t)dir * (Sn * Bn) * H3n;
#pragma unroll
    for (int v = 0; v < 8; ++v) {
        int m = v + 8 * half;
        int row = mt * 16 + m;
        gxd[(size_t)row * H3n + nt * 16 + idx] = f2bf_u16(acc[v] + bias);
    }
}

// ---------------- kernel 2: persistent GRU recurrence (one block per direction) ----------------
// Dynamic LDS layout (byte offsets within the 288KB allocation):
//   [0,       65536)  h_f32[64][256]
//   [65536,   98304)  h_bf16[64][256]
//   [98304,  196608)  gx slice buffer 0 (64 x 768 bf16)
//   [196608, 294912)  gx slice buffer 1
#define LDS_HF    0
#define LDS_HB    65536
#define LDS_GX0   98304
#define LDS_GXSZ  98304  // 64*768*2

// Issue async DMA of one gx step-slice (96KB) into LDS; 6 x B128 per lane.
__device__ __forceinline__ void async_load_slice(const unsigned short* gslice,
                                                 unsigned lds_base, int tid) {
#pragma unroll
    for (int j = 0; j < 6; ++j) {
        int chunk = j * 1024 + tid;                       // 6144 chunks of 16B
        unsigned lds_addr = lds_base + (unsigned)chunk * 16u;
        unsigned long long ga = (unsigned long long)(const void*)gslice
                              + (unsigned long long)chunk * 16ull;
        asm volatile("global_load_async_to_lds_b128 %0, %1, off"
                     :: "v"(lds_addr), "v"(ga)
                     : "memory");
    }
}

__global__ void gru_wmma_kernel(const unsigned short* __restrict__ gx,
                                const unsigned short* __restrict__ whh_bf,
                                const float* __restrict__ bhh_f, const float* __restrict__ bhh_b,
                                float* __restrict__ rnn_out) {
    extern __shared__ char smem[];
    float* h_f = (float*)(smem + LDS_HF);                   // [64][256] f32
    unsigned short* h_bu = (unsigned short*)(smem + LDS_HB);// [64][256] bf16

    int dir = blockIdx.x;
    const unsigned short* gxd = gx + (size_t)dir * (Sn * Bn) * H3n;
    const __bf16* whh = (const __bf16*)whh_bf + (size_t)dir * H3n * Hn;
    const float* bhh = dir ? bhh_b : bhh_f;

    int tid = threadIdx.x;
    for (int i = tid; i < Bn * Hn; i += blockDim.x) { h_f[i] = 0.f; h_bu[i] = 0; }

    // prologue: DMA first step's gx slice into buffer 0
    {
        int s0 = dir ? (Sn - 1) : 0;
        async_load_slice(gxd + (size_t)s0 * Bn * H3n, LDS_GX0, tid);
        asm volatile("s_wait_asynccnt 0x0" ::: "memory");
    }
    __syncthreads();

    int wave = tid >> 5;
    int lane = tid & 31;
    int idx = lane & 15, half = lane >> 4;

    for (int step = 0; step < Sn; ++step) {
        int s = dir ? (Sn - 1 - step) : step;
        unsigned gx_cur = LDS_GX0 + (unsigned)(step & 1) * LDS_GXSZ;
        unsigned gx_nxt = LDS_GX0 + (unsigned)((step + 1) & 1) * LDS_GXSZ;

        // DMA next step's slice while this step computes (buffer was consumed in step-1)
        if (step + 1 < Sn) {
            int sn = dir ? (Sn - 2 - step) : (step + 1);
            async_load_slice(gxd + (size_t)sn * Bn * H3n, gx_nxt, tid);
        }
        const unsigned short* gxlds = (const unsigned short*)(smem + gx_cur);

        float hnew[2][8];
#pragma unroll
        for (int ti = 0; ti < 2; ++ti) {
            int t = wave + ti * 32;        // 64 tiles: mt in [0,4), nt in [0,16)
            int mt = t >> 4, nt = t & 15;

            const __bf16* arow = (const __bf16*)h_bu + (size_t)(mt * 16 + idx) * Hn;
            const __bf16* brow_r = whh + (size_t)(0   + nt * 16 + idx) * Hn;
            const __bf16* brow_z = whh + (size_t)(256 + nt * 16 + idx) * Hn;
            const __bf16* brow_n = whh + (size_t)(512 + nt * 16 + idx) * Hn;

            f32x8 ar = {0.f,0.f,0.f,0.f,0.f,0.f,0.f,0.f};
            f32x8 az = {0.f,0.f,0.f,0.f,0.f,0.f,0.f,0.f};
            f32x8 an = {0.f,0.f,0.f,0.f,0.f,0.f,0.f,0.f};
#pragma unroll
            for (int kk = 0; kk < 8; ++kk) {
                int kb = kk * 32;
                bf16x16 a = load_a_bf16(arow, kb, half);
                ar = wmma_bf16(a, load_b_bf16(brow_r, kb, half), ar);
                az = wmma_bf16(a, load_b_bf16(brow_z, kb, half), az);
                an = wmma_bf16(a, load_b_bf16(brow_n, kb, half), an);
            }

            int n = idx;
            float br_ = bhh[nt * 16 + n];
            float bz_ = bhh[256 + nt * 16 + n];
            float bn_ = bhh[512 + nt * 16 + n];
#pragma unroll
            for (int v = 0; v < 8; ++v) {
                int m = v + 8 * half;
                int brow = mt * 16 + m;
                int col = nt * 16 + n;
                float xr = bfu2f(gxlds[brow * H3n + col]);
                float xz = bfu2f(gxlds[brow * H3n + 256 + col]);
                float xn = bfu2f(gxlds[brow * H3n + 512 + col]);
                float rg = dev_sigmoid(xr + ar[v] + br_);
                float zg = dev_sigmoid(xz + az[v] + bz_);
                float ng = tanhf(xn + rg * (an[v] + bn_));
                float hp = h_f[brow * Hn + col];
                hnew[ti][v] = (1.f - zg) * ng + zg * hp;
            }
        }
        __syncthreads();   // all reads of h done before overwrite
#pragma unroll
        for (int ti = 0; ti < 2; ++ti) {
            int t = wave + ti * 32;
            int mt = t >> 4, nt = t & 15;
#pragma unroll
            for (int v = 0; v < 8; ++v) {
                int m = v + 8 * half;
                int brow = mt * 16 + m;
                int col = nt * 16 + idx;
                float hv = hnew[ti][v];
                h_f[brow * Hn + col] = hv;
                h_bu[brow * Hn + col] = f2bf_u16(hv);
                rnn_out[((size_t)brow * Sn + s) * H2n + dir * Hn + col] = hv;
            }
        }
        // my async loads for the next slice must be in LDS before anyone reads them
        asm volatile("s_wait_asynccnt 0x0" ::: "memory");
        __syncthreads();   // publish h writes + async slice to all waves
    }
}

// ---------------- kernel 3: masked attention pooling ----------------
__global__ void attn_kernel(const float* __restrict__ rnn_out, const int* __restrict__ amask,
                            const float* __restrict__ att_w, const float* __restrict__ att_b,
                            float* __restrict__ doc) {
    __shared__ float sw[H2n];
    __shared__ float sc[Sn];
    __shared__ float red[256];
    int b = blockIdx.x, tid = threadIdx.x;
    for (int i = tid; i < H2n; i += 256) sw[i] = att_w[i];
    __syncthreads();
    float ab = att_b[0];
    for (int s = tid; s < Sn; s += 256) {
        const float* row = rnn_out + ((size_t)b * Sn + s) * H2n;
        float acc = 0.f;
        for (int h = 0; h < H2n; ++h) acc = fmaf(row[h], sw[h], acc);
        acc += ab;
        if (amask[b * Sn + s] == 0) acc = -1e30f;
        sc[s] = acc;
    }
    __syncthreads();
    float lmax = -1e30f;
    for (int s = tid; s < Sn; s += 256) lmax = fmaxf(lmax, sc[s]);
    red[tid] = lmax; __syncthreads();
    for (int off = 128; off > 0; off >>= 1) { if (tid < off) red[tid] = fmaxf(red[tid], red[tid + off]); __syncthreads(); }
    float smax = red[0]; __syncthreads();
    float lsum = 0.f;
    for (int s = tid; s < Sn; s += 256) { float e = __expf(sc[s] - smax); sc[s] = e; lsum += e; }
    red[tid] = lsum; __syncthreads();
    for (int off = 128; off > 0; off >>= 1) { if (tid < off) red[tid] += red[tid + off]; __syncthreads(); }
    float inv = 1.f / red[0]; __syncthreads();
    for (int h = tid; h < H2n; h += 256) {
        float acc = 0.f;
        for (int s = 0; s < Sn; ++s)
            acc = fmaf(sc[s], rnn_out[((size_t)b * Sn + s) * H2n + h], acc);
        doc[(size_t)b * H2n + h] = acc * inv;
    }
}

// ---------------- kernel 4: fc (+ optional eval-BN + relu) ----------------
__global__ void fc_kernel(const float* __restrict__ in, const float* __restrict__ w,
                          const float* __restrict__ bias, const float* __restrict__ gamma,
                          const float* __restrict__ beta, int din, int dout, int bn_relu,
                          float* __restrict__ out) {
    int row = blockIdx.x;
    const float* x = in + (size_t)row * din;
    const float inv_std = 0.99999500003750f; // 1/sqrt(1+1e-5)
    for (int j = threadIdx.x; j < dout; j += blockDim.x) {
        const float* wr = w + (size_t)j * din;
        float acc = bias[j];
        for (int k = 0; k < din; ++k) acc = fmaf(x[k], wr[k], acc);
        if (bn_relu) {
            acc = acc * inv_std * gamma[j] + beta[j];
            acc = fmaxf(acc, 0.f);
        }
        out[(size_t)row * dout + j] = acc;
    }
}

// ---------------- kernel 5: losses + preds ----------------
__global__ void loss_kernel(const float* __restrict__ logits, const int* __restrict__ labels,
                            const int* __restrict__ amask, float* __restrict__ out) {
    __shared__ float sprobs[Bn * Cn];
    __shared__ float red[256];
    __shared__ float rowloss[Bn];
    int tid = threadIdx.x;
    float ll = 0.f;
    for (int i = tid; i < Bn * Cn; i += 256) {
        float x = logits[i];
        float y = (float)labels[i];
        ll += fmaxf(x, 0.f) - x * y + log1pf(__expf(-fabsf(x)));
        float p = 1.f / (1.f + __expf(-x));
        sprobs[i] = p;
        out[2 + i] = (p > 0.3f) ? 1.f : 0.f;
    }
    red[tid] = ll; __syncthreads();
    for (int off = 128; off > 0; off >>= 1) { if (tid < off) red[tid] += red[tid + off]; __syncthreads(); }
    if (tid == 0) out[0] = red[0] / (float)(Bn * Cn);
    __syncthreads();
    if (tid < Bn) {
        int lens = 0;
        for (int s = 0; s < Sn; ++s) lens += amask[tid * Sn + s];
        int k = lens < 200 ? 2 : (lens < 400 ? 3 : 4);
        const float* p = sprobs + tid * Cn;
        float cur = 1e30f;
        for (int it = 0; it < k; ++it) {
            float best = -1e30f;
            for (int c = 0; c < Cn; ++c) { float v = p[c]; if (v < cur && v > best) best = v; }
            cur = best; // value of the (it+1)-th largest
        }
        float tsum = 0.f;
        for (int c = 0; c < Cn; ++c) {
            float y = (float)labels[tid * Cn + c];
            if (p[c] < cur) tsum += y * p[c];   // (1 - topk_mask)
        }
        rowloss[tid] = tsum;
    }
    __syncthreads();
    if (tid == 0) { float t = 0.f; for (int b = 0; b < Bn; ++b) t += rowloss[b]; out[1] = t / (float)Bn; }
}

// ---------------- host launch ----------------
extern "C" void kernel_launch(void* const* d_in, const int* in_sizes, int n_in,
                              void* d_out, int out_size, void* d_ws, size_t ws_size,
                              hipStream_t stream) {
    const int*   input_facts = (const int*)d_in[0];
    const int*   amask       = (const int*)d_in[2];
    const int*   law_labels  = (const int*)d_in[3];
    const float* emb    = (const float*)d_in[4];
    const float* W_ih_f = (const float*)d_in[5];
    const float* W_hh_f = (const float*)d_in[6];
    const float* b_ih_f = (const float*)d_in[7];
    const float* b_hh_f = (const float*)d_in[8];
    const float* W_ih_b = (const float*)d_in[9];
    const float* W_hh_b = (const float*)d_in[10];
    const float* b_ih_b = (const float*)d_in[11];
    const float* b_hh_b = (const float*)d_in[12];
    const float* att_w  = (const float*)d_in[13];
    const float* att_b  = (const float*)d_in[14];
    const float* w1 = (const float*)d_in[15]; const float* b1 = (const float*)d_in[16];
    const float* g1 = (const float*)d_in[17]; const float* be1 = (const float*)d_in[18];
    const float* w2 = (const float*)d_in[19]; const float* b2 = (const float*)d_in[20];
    const float* g2 = (const float*)d_in[21]; const float* be2 = (const float*)d_in[22];
    const float* w3 = (const float*)d_in[23]; const float* b3 = (const float*)d_in[24];
    const float* g3 = (const float*)d_in[25]; const float* be3 = (const float*)d_in[26];
    const float* w4 = (const float*)d_in[27]; const float* b4 = (const float*)d_in[28];

    // workspace layout (bytes)
    char* ws = (char*)d_ws;
    size_t off = 0;
    unsigned short* wih_bf = (unsigned short*)(ws + off); off += (size_t)2 * H3n * Hn * 2;   // 0.75MB
    unsigned short* whh_bf = (unsigned short*)(ws + off); off += (size_t)2 * H3n * Hn * 2;   // 0.75MB
    unsigned short* gx = (unsigned short*)(ws + off); off += (size_t)2 * Sn * Bn * H3n * 2;  // 96MB (bf16)
    float* rnn_out = (float*)(ws + off); off += (size_t)Bn * Sn * H2n * 4;                   // 64MB
    float* doc     = (float*)(ws + off); off += (size_t)Bn * H2n * 4;
    float* h1      = (float*)(ws + off); off += (size_t)Bn * MLPn * 4;
    float* h2      = (float*)(ws + off); off += (size_t)Bn * (MLPn / 2) * 4;
    float* h3      = (float*)(ws + off); off += (size_t)Bn * (MLPn / 4) * 4;
    float* logits  = (float*)(ws + off); off += (size_t)Bn * Cn * 4;
    (void)ws_size; (void)in_sizes; (void)n_in; (void)out_size;

    // 0: weights -> bf16
    {
        int total = 4 * H3n * Hn;
        cvt_bf16_kernel<<<(total + 255) / 256, 256, 0, stream>>>(
            W_ih_f, W_ih_b, W_hh_f, W_hh_b, wih_bf, whh_bf);
    }
    // 1: input projections via WMMA (both directions)
    {
        int tiles = 2 * ((Sn * Bn) / 16) * (H3n / 16); // 196608 wave-tiles
        gx_wmma_kernel<<<tiles / 8, 256, 0, stream>>>(
            input_facts, emb, wih_bf, b_ih_f, b_ih_b, gx);
    }
    // 2: persistent GRU recurrence with async-LDS double-buffered gx slices
    {
        size_t lds = (size_t)LDS_GX0 + 2 * (size_t)LDS_GXSZ;  // 288KB
        gru_wmma_kernel<<<2, 1024, lds, stream>>>(gx, whh_bf, b_hh_f, b_hh_b, rnn_out);
    }
    // 3: attention pooling
    attn_kernel<<<Bn, 256, 0, stream>>>(rnn_out, amask, att_w, att_b, doc);
    // 4: MLP head
    fc_kernel<<<Bn, 256, 0, stream>>>(doc, w1, b1, g1, be1, H2n, MLPn, 1, h1);
    fc_kernel<<<Bn, 256, 0, stream>>>(h1, w2, b2, g2, be2, MLPn, MLPn / 2, 1, h2);
    fc_kernel<<<Bn, 256, 0, stream>>>(h2, w3, b3, g3, be3, MLPn / 2, MLPn / 4, 1, h3);
    fc_kernel<<<Bn, 256, 0, stream>>>(h3, w4, b4, nullptr, nullptr, MLPn / 4, Cn, 0, logits);
    // 5: losses + predictions
    loss_kernel<<<1, 256, 0, stream>>>(logits, law_labels, amask, (float*)d_out);
}
